// FNO_16879221473377
// MI455X (gfx1250) — matrix operand order repaired and checked by reading, not compiled, not run on hardware
//
#include <hip/hip_runtime.h>
#include <math.h>

typedef _Float16 h16;
typedef __attribute__((ext_vector_type(16))) _Float16 v16h;
typedef __attribute__((ext_vector_type(8)))  float    v8f;

#define BB 16
#define HH 256
#define WW 256
#define DD 32
#define LL 4
#define MM 12
#define HWP   (HH*WW)            // 65536 pixels per image
#define NPIX  ((size_t)BB*HWP)   // 1048576
#define ROWS  (BB*DD*HH)         // 131072 rows for the W-axis GEMMs
#define TILES (ROWS/16)          // 8192 16-row M-tiles

// ---------------- workspace layout (bytes, all 256B aligned) ----------------
#define SZ_HF   ((size_t)BB*DD*HH*WW*4)          // 134,217,728 (one activation buffer)
#define OFF_HA  ((size_t)0)
#define OFF_HB  (OFF_HA + SZ_HF)
#define OFF_H1  (OFF_HB + SZ_HF)                 // [ROWS][24] f32 partial DFT (W-axis)
#define SZ_H1   ((size_t)ROWS*24*4)
#define OFF_FT  (OFF_H1 + SZ_H1)                 // [B*D][24][12][2] f32 spectral modes
#define SZ_FT   ((size_t)BB*DD*24*12*2*4)
#define OFF_G   (OFF_FT + SZ_FT)                 // mixed modes, same shape
#define OFF_Y1  (OFF_G + SZ_FT)                  // [TILES][32 lane][16 e] f16 A-swizzled y1
#define SZ_Y1   ((size_t)TILES*512*2)
#define OFF_FW  (OFF_Y1 + SZ_Y1)                 // fwd W-basis, B-frag swizzled: [8kt*2nt][32][16] f16
#define SZ_FW   ((size_t)8*2*512*2)
#define OFF_BI  (OFF_FW + SZ_FW)                 // inv W-basis: [16nt][32][16] f16
#define SZ_BI   ((size_t)16*512*2)
#define OFF_LW  (OFF_BI + SZ_BI)                 // lw per layer, B-frag swizzled: [L*2nt][32][16] f16
#define SZ_LW   ((size_t)LL*2*512*2)
#define OFF_E   (OFF_LW + SZ_LW)                 // fwd H-basis [24 k1][256 x][2] f32
#define SZ_E    ((size_t)24*256*2*4)
#define OFF_EI  (OFF_E + SZ_E)                   // inv H-basis [256 x][24 k1][2] f32
#define SZ_EI   ((size_t)256*24*2*4)
#define OFF_ST  (OFF_EI + SZ_EI)                 // BN partial stats [L][32][{sum,sumsq}] f32
#define SZ_ST   ((size_t)LL*64*4)
#define OFF_SS  (OFF_ST + SZ_ST)                 // BN scale/shift slots [L+1][32][{sc,sh}] f32
#define SZ_SS   ((size_t)(LL+1)*64*4)
// total ~ 280 MB in d_ws

#define WMMA(a,b,c) __builtin_amdgcn_wmma_f32_16x16x32_f16(false,(a),false,(b),(short)0,(c),false,false)

// ---------------------------------------------------------------------------
// init: build swizzled DFT bases + swizzled lw (f16), H-axis bases (f32),
// zero BN stats, identity affine for layer 0. Runs every call (deterministic).
// regions: [0,8192) Fw | [8192,16384) BI | [16384,20480) lw | [20480,26624) E
//          [26624,32768) EI | [32768,33024) stats=0 | [33024,33088) ss0=identity
__global__ void k_init(const float* __restrict__ lw, char* __restrict__ ws) {
  int t = blockIdx.x * 256 + threadIdx.x;
  const float PI2 = 6.283185307179586f;
  h16*  fw  = (h16*)(ws + OFF_FW);
  h16*  bi  = (h16*)(ws + OFF_BI);
  h16*  lws = (h16*)(ws + OFF_LW);
  float* E  = (float*)(ws + OFF_E);
  float* EI = (float*)(ws + OFF_EI);
  float* st = (float*)(ws + OFF_ST);
  float* ss = (float*)(ws + OFF_SS);
  if (t < 8192) {                       // forward W basis (cos | -sin), B-frag order
    int e = t & 15, lane = (t >> 4) & 31, nt = (t >> 9) & 1, kt = t >> 10;
    int k = (e & 7) + 8 * (lane >> 4) + 16 * (e >> 3);       // B-frag row (K)
    int n = nt * 16 + (lane & 15);      // B-frag col (N)
    int y = kt * 32 + k;
    float v = 0.f;
    if (n < 12)      v =  cosf(PI2 * (float)((y * n) & 255) * (1.f/256.f));
    else if (n < 24) v = -sinf(PI2 * (float)((y * (n-12)) & 255) * (1.f/256.f));
    fw[t] = (h16)v;
  } else if (t < 16384) {               // inverse W basis (w*cos | -w*sin)/(H*W)
    int t2 = t - 8192;
    int e = t2 & 15, lane = (t2 >> 4) & 31, nt = t2 >> 9;
    int k = (e & 7) + 8 * (lane >> 4) + 16 * (e >> 3);
    int yq = nt * 16 + (lane & 15);
    float v = 0.f;
    const float inv = 1.f / (float)(HH * WW);
    if (k < 12) {
      float w = (k == 0) ? 1.f : 2.f;
      v =  w * inv * cosf(PI2 * (float)((k * yq) & 255) * (1.f/256.f));
    } else if (k < 24) {
      int kk = k - 12;
      float w = (kk == 0) ? 1.f : 2.f;
      v = -w * inv * sinf(PI2 * (float)((kk * yq) & 255) * (1.f/256.f));
    }
    bi[t2] = (h16)v;
  } else if (t < 20480) {               // lw^T per layer, B-frag order, f16
    int t3 = t - 16384;
    int e = t3 & 15, lane = (t3 >> 4) & 31, lnt = t3 >> 9;
    int l = lnt >> 1, nt = lnt & 1;
    int c = (e & 7) + 8 * (lane >> 4) + 16 * (e >> 3);               // K = input channel
    int op = nt * 16 + (lane & 15);             // N = output channel
    lws[t3] = (h16)lw[((size_t)(l * 32 + op)) * 32 + c];
  } else if (t < 26624) {               // forward H basis e^{-2pi i x k1'/H}
    int t4 = t - 20480;
    int k1 = t4 >> 8, x = t4 & 255;
    int k1p = (k1 < 12) ? k1 : (256 - 24 + k1);
    float ang = -PI2 * (float)((x * k1p) & 255) * (1.f/256.f);
    E[t4*2] = cosf(ang); E[t4*2+1] = sinf(ang);
  } else if (t < 32768) {               // inverse H basis e^{+2pi i x k1'/H}, [x][k1]
    int t5 = t - 26624;
    int x = t5 / 24, k1 = t5 % 24;
    int k1p = (k1 < 12) ? k1 : (256 - 24 + k1);
    float ang = PI2 * (float)((x * k1p) & 255) * (1.f/256.f);
    EI[(x*24 + k1)*2] = cosf(ang); EI[(x*24 + k1)*2+1] = sinf(ang);
  } else if (t < 33024) {               // zero BN partial stats
    st[t - 32768] = 0.f;
  } else if (t < 33088) {               // identity affine for layer 0 load
    int i = t - 33024;
    ss[i] = (i & 1) ? 0.f : 1.f;
  }
}

// ---------------------------------------------------------------------------
// linear_in: h[b,d,x,y] = w_in[d,:]·[x, cx0, cx1] + b_in[d]
__global__ void k_linear_in(const float* __restrict__ x, const float* __restrict__ cx,
                            const float* __restrict__ w_in, const float* __restrict__ b_in,
                            float* __restrict__ h) {
  size_t i = (size_t)blockIdx.x * 256 + threadIdx.x;
  int p = (int)(i & (HWP - 1));
  int d = (int)((i >> 16) & (DD - 1));
  int b = (int)(i >> 21);
  h[i] = w_in[d*3+0] * x[((size_t)b << 16) + p]
       + w_in[d*3+1] * cx[p]
       + w_in[d*3+2] * cx[HWP + p]
       + b_in[d];
}

// ---------------------------------------------------------------------------
// W-axis partial DFT as WMMA GEMM: [16 rows x 256] x [256 x 24(pad 32)].
// A-fragments are gathered DIRECTLY from global memory: the ISA f16 16x32
// A layout needs, per lane, two contiguous 32B runs of its row
// (k = 8*hi+0..7 and k = 16+8*hi+0..7) -> four aligned b128 loads, no LDS,
// no barriers. The 16KB per-wave working set stays resident across the 8
// K-steps, so full cachelines are consumed. Applies the previous layer's
// folded BN affine (scale,shift per channel) on load.
__global__ void __launch_bounds__(256) k_dft_w(const float* __restrict__ hin,
                                               const float* __restrict__ ss,
                                               const h16* __restrict__ fw,
                                               float* __restrict__ h1) {
  int lane = threadIdx.x & 31, wv = threadIdx.x >> 5;
  int lr = lane & 15, lh = lane >> 4;
  int tile = blockIdx.x * 8 + wv;
  int row0 = tile * 16;                 // rows are (b,d,x); 16 | H so tile stays in one (b,d)
  int d = (row0 >> 8) & 31;
  float sc = ss[d*2], sh = ss[d*2+1];
  const float* row = hin + (size_t)(row0 + lr) * WW + 8 * lh;  // this lane's row, hi-half base
  v8f acc0 = {}; v8f acc1 = {};
  for (int kt = 0; kt < 8; ++kt) {
    const float* g = row + kt * 32;
    float4 f0 = *(const float4*)(g + 0);     // k = 8*lh + 0..3
    float4 f1 = *(const float4*)(g + 4);     // k = 8*lh + 4..7
    float4 f2 = *(const float4*)(g + 16);    // k = 16 + 8*lh + 0..3
    float4 f3 = *(const float4*)(g + 20);    // k = 16 + 8*lh + 4..7
    if (kt < 7) __builtin_prefetch(g + 32, 0, 1);   // global_prefetch_b8 next K-tile
    v16h a;
    a[0]  = (h16)(f0.x * sc + sh); a[1]  = (h16)(f0.y * sc + sh);
    a[2]  = (h16)(f0.z * sc + sh); a[3]  = (h16)(f0.w * sc + sh);
    a[4]  = (h16)(f1.x * sc + sh); a[5]  = (h16)(f1.y * sc + sh);
    a[6]  = (h16)(f1.z * sc + sh); a[7]  = (h16)(f1.w * sc + sh);
    a[8]  = (h16)(f2.x * sc + sh); a[9]  = (h16)(f2.y * sc + sh);
    a[10] = (h16)(f2.z * sc + sh); a[11] = (h16)(f2.w * sc + sh);
    a[12] = (h16)(f3.x * sc + sh); a[13] = (h16)(f3.y * sc + sh);
    a[14] = (h16)(f3.z * sc + sh); a[15] = (h16)(f3.w * sc + sh);
    const h16* fb = fw + (size_t)(kt * 2 * 32 + lane) * 16;
    v16h b0 = *(const v16h*)fb;
    v16h b1 = *(const v16h*)(fb + 512);
    acc0 = WMMA(a, b0, acc0);
    acc1 = WMMA(a, b1, acc1);
  }
#pragma unroll
  for (int v = 0; v < 8; ++v) {
    int r = row0 + v + 8 * lh;                      // D-frag m = v + 8*hi
    h1[(size_t)r * 24 + lr] = acc0[v];
    if (lr < 8) h1[(size_t)r * 24 + 16 + lr] = acc1[v];
  }
}

// ---------------------------------------------------------------------------
// H-axis DFT (24 retained k1 rows): complex dot of length 256 per (b,d,k1,k2).
__global__ void k_dft_h(const float* __restrict__ h1, const float* __restrict__ E,
                        float* __restrict__ ft) {
  int i = blockIdx.x * 256 + threadIdx.x;
  int k2 = i % 12, k1 = (i / 12) % 24, bd = i / 288;
  const float* hp = h1 + (size_t)bd * 256 * 24 + k2;
  const float* ep = E + k1 * 512;
  float ar = 0.f, ai = 0.f;
  for (int x = 0; x < 256; ++x) {
    float er = ep[2*x], ei = ep[2*x+1];
    float hr = hp[x*24], hi = hp[x*24+12];
    ar += er * hr - ei * hi;
    ai += er * hi + ei * hr;
  }
  size_t o = (size_t)bd * 576 + k1 * 24 + k2 * 2;
  ft[o] = ar; ft[o+1] = ai;
}

// ---------------------------------------------------------------------------
// per-mode complex channel mix: G[b,o,k1,k2] = sum_i ft[b,i,k1,k2]*w[i,o,..]
__global__ void k_mix(const float* __restrict__ ft,
                      const float* __restrict__ w1r, const float* __restrict__ w1i,
                      const float* __restrict__ w2r, const float* __restrict__ w2i,
                      int layer, float* __restrict__ G) {
  int i = blockIdx.x * 256 + threadIdx.x;
  int k2 = i % 12, k1 = (i / 12) % 24, o = (i / 288) % 32, b = i / 9216;
  int top = (k1 < 12);
  int km = top ? k1 : k1 - 12;
  size_t wbase = ((size_t)layer * 1024 + o) * 144 + km * 12 + k2;
  const float* wr = (top ? w1r : w2r) + wbase;
  const float* wi = (top ? w1i : w2i) + wbase;
  const float* fp = ft + (size_t)b * 32 * 576 + k1 * 24 + k2 * 2;
  float ar = 0.f, ai = 0.f;
  for (int c = 0; c < 32; ++c) {
    float fr = fp[c*576], fi = fp[c*576+1];
    float xr = wr[(size_t)c*4608], xi = wi[(size_t)c*4608];
    ar += fr * xr - fi * xi;
    ai += fr * xi + fi * xr;
  }
  size_t oo = (size_t)(b * 32 + o) * 576 + k1 * 24 + k2 * 2;
  G[oo] = ar; G[oo+1] = ai;
}

// ---------------------------------------------------------------------------
// inverse H-axis (24-term complex dot), result stored f16 pre-swizzled into
// WMMA A-fragment order (K = [12 real | 12 imag | 8 zero pad]).
__global__ void k_idft_h(const float* __restrict__ G, const float* __restrict__ EI,
                         h16* __restrict__ y1) {
  int i = blockIdx.x * 256 + threadIdx.x;
  int j = i & 15;
  int row = i >> 4;
  int x = row & 255, bd = row >> 8;
  int tile = row >> 4, m = row & 15;
  size_t tb = (size_t)tile * 512;
  if (j < 12) {
    const float* gp = G + (size_t)bd * 576 + j * 2;
    const float* ep = EI + x * 48;
    float ar = 0.f, ai = 0.f;
#pragma unroll
    for (int k1 = 0; k1 < 24; ++k1) {
      float gr = gp[k1*24], gi = gp[k1*24+1];
      float er = ep[2*k1], es = ep[2*k1+1];
      ar += gr * er - gi * es;
      ai += gr * es + gi * er;
    }
    int k = j;       { int ln = m + 16*((k>>3)&1); int e = 8*(k>>4) + (k&7); y1[tb + ln*16 + e] = (h16)ar; }
    k = 12 + j;      { int ln = m + 16*((k>>3)&1); int e = 8*(k>>4) + (k&7); y1[tb + ln*16 + e] = (h16)ai; }
  } else {           // zero padding K = 24..31
    int k = 12 + j;  { int ln = m + 16*((k>>3)&1); int e = 8*(k>>4) + (k&7); y1[tb + ln*16 + e] = (h16)0.f; }
    k = 16 + j;      { int ln = m + 16*((k>>3)&1); int e = 8*(k>>4) + (k&7); y1[tb + ln*16 + e] = (h16)0.f; }
  }
}

// ---------------------------------------------------------------------------
// FUSED: inverse W-axis WMMA GEMM (all 32 channels of a 16x16 pixel patch)
//        -> LDS -> residual pointwise WMMA (y + y*lw^T + lb) -> ReLU
//        -> BN partial stats (ds atomics) -> coalesced store of z.
__global__ void __launch_bounds__(256) k_idft_w_fused(const h16* __restrict__ y1,
                                                      const h16* __restrict__ bi,
                                                      const h16* __restrict__ lws,
                                                      const float* __restrict__ lb,
                                                      float* __restrict__ z,
                                                      float* __restrict__ stl) {
  __shared__ float lds[256 * 33];        // [pixel][channel], pitch 33 vs bank conflicts
  __shared__ float sred[64];
  int lane = threadIdx.x & 31, wv = threadIdx.x >> 5;
  int lr = lane & 15, lh = lane >> 4;
  int yt = blockIdx.x, xt = blockIdx.y, b = blockIdx.z;
  if (threadIdx.x < 64) sred[threadIdx.x] = 0.f;

  // phase 1: y = y1 * BI for this block's 16x16 patch, 4 channels per wave
  v16h bfr = *(const v16h*)(bi + (size_t)(yt * 32 + lane) * 16);
  v8f acc[4];
#pragma unroll
  for (int j = 0; j < 4; ++j) {
    int o = wv * 4 + j;
    int tile = (b * 32 + o) * 16 + xt;
    v16h a = *(const v16h*)(y1 + (size_t)(tile * 32 + lane) * 16);
    v8f c = {};
    acc[j] = WMMA(a, bfr, c);
  }
#pragma unroll
  for (int j = 0; j < 4; ++j) {
    int o = wv * 4 + j;
#pragma unroll
    for (int v = 0; v < 8; ++v) {
      int px = v + 8 * lh;               // row within patch (x_local)
      lds[(px * 16 + lr) * 33 + o] = acc[j][v];
    }
  }
  __syncthreads();

  // phase 2: pointwise channel mix over K=32, residual + bias + ReLU + stats
  v16h bw0 = *(const v16h*)(lws + (size_t)lane * 16);
  v16h bw1 = *(const v16h*)(lws + (size_t)(32 + lane) * 16);
  float lb0 = lb[lr], lb1 = lb[16 + lr];
  float s0 = 0.f, q0 = 0.f, s1 = 0.f, q1 = 0.f;
#pragma unroll
  for (int pt = 0; pt < 2; ++pt) {
    int ptile = wv * 2 + pt;             // 16 pixel-tiles over 8 waves
    v16h a;
#pragma unroll
    for (int e = 0; e < 16; ++e) {
      int k = 16 * (e >> 3) + (e & 7) + 8 * lh;
      a[e] = (h16)lds[(ptile * 16 + lr) * 33 + k];
    }
    v8f c0 = {}; v8f c1 = {};
    c0 = WMMA(a, bw0, c0);
    c1 = WMMA(a, bw1, c1);
#pragma unroll
    for (int v = 0; v < 8; ++v) {
      int p = ptile * 16 + v + 8 * lh;
      float y0 = lds[p * 33 + lr];
      float z0 = y0 + c0[v] + lb0; z0 = z0 > 0.f ? z0 : 0.f;
      lds[p * 33 + lr] = z0; s0 += z0; q0 += z0 * z0;
      float y1v = lds[p * 33 + 16 + lr];
      float z1 = y1v + c1[v] + lb1; z1 = z1 > 0.f ? z1 : 0.f;
      lds[p * 33 + 16 + lr] = z1; s1 += z1; q1 += z1 * z1;
    }
  }
  atomicAdd(&sred[lr * 2],        s0); atomicAdd(&sred[lr * 2 + 1],        q0);
  atomicAdd(&sred[(16+lr) * 2],   s1); atomicAdd(&sred[(16+lr) * 2 + 1],   q1);
  __syncthreads();

  // coalesced store of z (channel-planar [B,D,H,W]) from LDS transpose
#pragma unroll
  for (int it = 0; it < 32; ++it) {
    int flat = threadIdx.x + it * 256;
    int o = flat >> 8, p = flat & 255;
    int px = p >> 4, py = p & 15;
    z[((size_t)((b * 32 + o) * 256 + xt * 16 + px) << 8) + yt * 16 + py] = lds[p * 33 + o];
  }
  if (threadIdx.x < 64) atomicAdd(&stl[threadIdx.x], sred[threadIdx.x]);
}

// ---------------------------------------------------------------------------
// finalize BN: fold mean/var/gamma/beta into per-channel (scale, shift) that
// the next layer (or linear_out) applies on load.
__global__ void k_bn_finalize(const float* __restrict__ stl, const float* __restrict__ gamma,
                              const float* __restrict__ beta, int layer,
                              float* __restrict__ ssout) {
  int d = threadIdx.x;
  if (d >= 32) return;
  float n = (float)NPIX;
  float mean = stl[d*2] / n;
  float var = stl[d*2+1] / n - mean * mean;
  float rstd = rsqrtf(var + 1e-5f);
  float sc = gamma[layer*32 + d] * rstd;
  ssout[d*2] = sc;
  ssout[d*2+1] = beta[layer*32 + d] - mean * sc;
}

// ---------------------------------------------------------------------------
// linear_out with layer-3 BN affine folded in.
__global__ void k_linear_out(const float* __restrict__ h, const float* __restrict__ ss,
                             const float* __restrict__ w_out, const float* __restrict__ b_out,
                             float* __restrict__ out) {
  size_t i = (size_t)blockIdx.x * 256 + threadIdx.x;
  int p = (int)(i & (HWP - 1));
  int b = (int)(i >> 16);
  float acc = b_out[0];
#pragma unroll
  for (int d = 0; d < 32; ++d) {
    float v = h[((size_t)(b * 32 + d) << 16) + p];
    acc += (v * ss[d*2] + ss[d*2+1]) * w_out[d];
  }
  out[i] = acc;
}

// ---------------------------------------------------------------------------
extern "C" void kernel_launch(void* const* d_in, const int* in_sizes, int n_in,
                              void* d_out, int out_size, void* d_ws, size_t ws_size,
                              hipStream_t stream) {
  const float* x     = (const float*)d_in[0];
  const float* cx    = (const float*)d_in[1];
  const float* w_in  = (const float*)d_in[2];
  const float* b_in  = (const float*)d_in[3];
  const float* w1r   = (const float*)d_in[4];
  const float* w1i   = (const float*)d_in[5];
  const float* w2r   = (const float*)d_in[6];
  const float* w2i   = (const float*)d_in[7];
  const float* lw    = (const float*)d_in[8];
  const float* lb    = (const float*)d_in[9];
  const float* gamma = (const float*)d_in[10];
  const float* beta  = (const float*)d_in[11];
  const float* w_out = (const float*)d_in[12];
  const float* b_out = (const float*)d_in[13];

  char* ws = (char*)d_ws;
  float* hA  = (float*)(ws + OFF_HA);
  float* hB  = (float*)(ws + OFF_HB);
  float* h1  = (float*)(ws + OFF_H1);
  float* ft  = (float*)(ws + OFF_FT);
  float* G   = (float*)(ws + OFF_G);
  h16*   y1  = (h16*)(ws + OFF_Y1);
  h16*   fw  = (h16*)(ws + OFF_FW);
  h16*   bi  = (h16*)(ws + OFF_BI);
  h16*   lws = (h16*)(ws + OFF_LW);
  float* E   = (float*)(ws + OFF_E);
  float* EI  = (float*)(ws + OFF_EI);
  float* st  = (float*)(ws + OFF_ST);
  float* ssb = (float*)(ws + OFF_SS);

  k_init<<<130, 256, 0, stream>>>(lw, ws);
  k_linear_in<<<(unsigned)((NPIX * DD) / 256), 256, 0, stream>>>(x, cx, w_in, b_in, hA);
  for (int l = 0; l < LL; ++l) {
    float* hin  = (l & 1) ? hB : hA;
    float* hout = (l & 1) ? hA : hB;
    k_dft_w<<<TILES / 8, 256, 0, stream>>>(hin, ssb + l * 64, fw, h1);
    k_dft_h<<<(BB * DD * 288) / 256, 256, 0, stream>>>(h1, E, ft);
    k_mix<<<(BB * DD * 288) / 256, 256, 0, stream>>>(ft, w1r, w1i, w2r, w2i, l, G);
    k_idft_h<<<(ROWS * 16) / 256, 256, 0, stream>>>(G, EI, y1);
    k_idft_w_fused<<<dim3(WW / 16, HH / 16, BB), 256, 0, stream>>>(
        y1, bi, lws + (size_t)l * 1024, lb + l * 32, hout, st + l * 64);
    k_bn_finalize<<<1, 32, 0, stream>>>(st + l * 64, gamma, beta, l, ssb + (l + 1) * 64);
  }
  k_linear_out<<<(unsigned)(NPIX / 256), 256, 0, stream>>>(hA, ssb + LL * 64, w_out, b_out,
                                                           (float*)d_out);
}
